// MultiLatentAttention_10660108829033
// MI455X (gfx1250) — compile-verified, matrix operand order checked by reading, not run on hardware
//
#include <hip/hip_runtime.h>
#include <cstdint>
#include <cstddef>

// ---------------------------------------------------------------------------
// Types for CDNA5 WMMA (wave32): v_wmma_f32_16x16x32_bf16
// ---------------------------------------------------------------------------
typedef __attribute__((ext_vector_type(16))) __bf16 bf16x16;
typedef __attribute__((ext_vector_type(8)))  __bf16 bf16x8;
typedef __attribute__((ext_vector_type(8)))  float  f32x8;

#define S_LEN   2048
#define HID_DIM 2048
#define NHEAD   16
#define QD_DIM  128          // nope(64) + rope(64)
#define KVD_DIM 192          // k_nope(64) + v(128)
#define KLR_DIM 512
#define ATT_SCALE 0.08838834764831845f   // 128^-0.5
#define NEG_BIG (-3.0e38f)

// Native bf16 conversion (RTE) — lowers to v_cvt_pk_bf16_f32.
__device__ __forceinline__ unsigned short f32_to_bf16(float f) {
  __bf16 b = (__bf16)f;
  return __builtin_bit_cast(unsigned short, b);
}

__device__ __forceinline__ void store4_bf16(unsigned short* p, float4 v) {
  union { __bf16 b[4]; uint2 d; } pk;
  pk.b[0] = (__bf16)v.x; pk.b[1] = (__bf16)v.y;
  pk.b[2] = (__bf16)v.z; pk.b[3] = (__bf16)v.w;
  *reinterpret_cast<uint2*>(p) = pk.d;       // 8B LDS store
}

// Pack 4 independent floats -> 4 bf16 -> one 8B LDS store (transpose path).
__device__ __forceinline__ void store4_bf16v(unsigned short* p,
                                             float a, float b, float c, float d) {
  union { __bf16 b[4]; uint2 d2; } pk;
  pk.b[0] = (__bf16)a; pk.b[1] = (__bf16)b;
  pk.b[2] = (__bf16)c; pk.b[3] = (__bf16)d;
  *reinterpret_cast<uint2*>(p) = pk.d2;
}

// ISA 16-bit A/B fragment: per lane, two contiguous 8-element (16B) K-chunks.
__device__ __forceinline__ bf16x16 load_frag(const unsigned short* p0,
                                             const unsigned short* p1) {
  bf16x8 lo = *reinterpret_cast<const bf16x8*>(p0);
  bf16x8 hi = *reinterpret_cast<const bf16x8*>(p1);
  return __builtin_shufflevector(lo, hi, 0,1,2,3,4,5,6,7,8,9,10,11,12,13,14,15);
}

__device__ __forceinline__ f32x8 wmma_bf16(bf16x16 a, bf16x16 b, f32x8 c) {
  return __builtin_amdgcn_wmma_f32_16x16x32_bf16(false, a, false, b,
                                                 (short)0, c, false, false);
}

// ---------------------------------------------------------------------------
// Generic GEMM: C[M,N] = A[M,K] @ B[K,N], fp32 in/out, bf16 WMMA inside.
// Block tile 128x128, BK=32, 256 threads = 8 waves (2x4), wave tile 64x32.
// Register double-buffer across K-tiles; B transposed via 4x4 register blocks
// so the staging path is packed ds_store_b64 instead of scalar b16 stores.
// ---------------------------------------------------------------------------
#define G_BM 128
#define G_BN 128
#define G_BK 32
#define G_LDA 40   // 32 + 8 pad (ushorts): rows start on distinct bank groups

__global__ __launch_bounds__(256)
void gemm_bf16_kernel(const float* __restrict__ A, const float* __restrict__ B,
                      float* __restrict__ C, int M, int N, int K) {
  const int t    = threadIdx.x;
  const int w    = t >> 5;
  const int lane = t & 31;
  const int l16  = lane & 15;
  const int half = lane >> 4;
  const int wm   = w >> 2;   // 0..1
  const int wn   = w & 3;    // 0..3
  const int m0   = blockIdx.y * G_BM;
  const int n0   = blockIdx.x * G_BN;

  __shared__ __align__(16) unsigned short As[G_BM * G_LDA];
  __shared__ __align__(16) unsigned short Bs[G_BN * G_LDA];  // transposed [n][k]

  f32x8 acc[4][2];
#pragma unroll
  for (int mt = 0; mt < 4; ++mt)
#pragma unroll
    for (int nt = 0; nt < 2; ++nt)
#pragma unroll
      for (int r = 0; r < 8; ++r) acc[mt][nt][r] = 0.0f;

  // A staging coords: 4 float4 per thread, row-major contiguous
  auto load_A = [&](int k0, int i) -> float4 {
    int linear = (t + i * 256) * 4;
    int r = linear >> 5, c = linear & 31;
    return *reinterpret_cast<const float4*>(&A[(size_t)(m0 + r) * K + k0 + c]);
  };
  // B staging: each thread owns a 4x4 block (kq, nq) of the 32x128 tile
  const int kq = (t >> 5) * 4;      // 0..28
  const int nq = (t & 31) * 4;      // 0..124
  const int gn = n0 + nq;
  auto load_B = [&](int k0, float4 (&vb)[4]) {
#pragma unroll
    for (int j = 0; j < 4; ++j) {
      vb[j] = make_float4(0.f, 0.f, 0.f, 0.f);
      if (gn < N)
        vb[j] = *reinterpret_cast<const float4*>(
            &B[(size_t)(k0 + kq + j) * N + gn]);
    }
  };

  float4 pa[4], pb[4];
#pragma unroll
  for (int i = 0; i < 4; ++i) pa[i] = load_A(0, i);
  load_B(0, pb);

  for (int k0 = 0; k0 < K; k0 += G_BK) {
    // ---- commit staged registers to LDS (convert f32 -> bf16) ----
#pragma unroll
    for (int i = 0; i < 4; ++i) {
      int linear = (t + i * 256) * 4;
      int r = linear >> 5, c = linear & 31;
      store4_bf16(&As[r * G_LDA + c], pa[i]);
    }
    // 4x4 register transpose: 4 packed 8B stores
    store4_bf16v(&Bs[(nq + 0) * G_LDA + kq], pb[0].x, pb[1].x, pb[2].x, pb[3].x);
    store4_bf16v(&Bs[(nq + 1) * G_LDA + kq], pb[0].y, pb[1].y, pb[2].y, pb[3].y);
    store4_bf16v(&Bs[(nq + 2) * G_LDA + kq], pb[0].z, pb[1].z, pb[2].z, pb[3].z);
    store4_bf16v(&Bs[(nq + 3) * G_LDA + kq], pb[0].w, pb[1].w, pb[2].w, pb[3].w);
    __syncthreads();

    // ---- prefetch next K-tile: loads overlap the WMMAs below ----
    if (k0 + G_BK < K) {
#pragma unroll
      for (int i = 0; i < 4; ++i) pa[i] = load_A(k0 + G_BK, i);
      load_B(k0 + G_BK, pb);
    }

    // ---- WMMA on current tile ----
    bf16x16 af[4];
#pragma unroll
    for (int mt = 0; mt < 4; ++mt) {
      const int row = wm * 64 + mt * 16 + l16;
      af[mt] = load_frag(&As[row * G_LDA + half * 8],
                         &As[row * G_LDA + 16 + half * 8]);
    }
#pragma unroll
    for (int nt = 0; nt < 2; ++nt) {
      const int col = wn * 32 + nt * 16 + l16;
      bf16x16 bfr = load_frag(&Bs[col * G_LDA + half * 8],
                              &Bs[col * G_LDA + 16 + half * 8]);
#pragma unroll
      for (int mt = 0; mt < 4; ++mt)
        acc[mt][nt] = wmma_bf16(af[mt], bfr, acc[mt][nt]);
    }
    __syncthreads();
  }

#pragma unroll
  for (int mt = 0; mt < 4; ++mt)
#pragma unroll
    for (int nt = 0; nt < 2; ++nt) {
      int col = n0 + wn * 32 + nt * 16 + l16;
      if (col < N) {
#pragma unroll
        for (int r = 0; r < 8; ++r) {
          int row = m0 + wm * 64 + mt * 16 + 8 * half + r;
          C[(size_t)row * N + col] = acc[mt][nt][r];
        }
      }
    }
}

// ---------------------------------------------------------------------------
// RMSNorm over KLR=512 latent: one row per block, 256 threads.
// ---------------------------------------------------------------------------
__global__ __launch_bounds__(256)
void rmsnorm_kernel(const float* __restrict__ ckv_full,
                    const float* __restrict__ w, float* __restrict__ out) {
  const int row = blockIdx.x, t = threadIdx.x;
  float x0 = ckv_full[(size_t)row * 576 + t];
  float x1 = ckv_full[(size_t)row * 576 + 256 + t];
  __shared__ float red[256];
  red[t] = x0 * x0 + x1 * x1;
  __syncthreads();
  for (int s = 128; s > 0; s >>= 1) {
    if (t < s) red[t] += red[t + s];
    __syncthreads();
  }
  float scale = rsqrtf(red[0] * (1.0f / 512.0f) + 1e-6f);
  out[(size_t)row * 512 + t]       = x0 * scale * w[t];
  out[(size_t)row * 512 + 256 + t] = x1 * scale * w[256 + t];
}

// ---------------------------------------------------------------------------
// Interleaved RoPE: q_pe (in-place in q_buf) + k_pe -> k_pe_rot.
// ---------------------------------------------------------------------------
__global__ __launch_bounds__(256)
void rope_kernel(float* __restrict__ q_buf, const float* __restrict__ ckv_full,
                 float* __restrict__ k_pe_rot, const float* __restrict__ cosb,
                 const float* __restrict__ sinb) {
  const int row = blockIdx.x, t = threadIdx.x;
  const float* cr = cosb + (size_t)row * 64;
  const float* sr = sinb + (size_t)row * 64;

  float o0[2], o1[2];
  int hh[2], jj[2];
#pragma unroll
  for (int it = 0; it < 2; ++it) {
    int idx = t + it * 256;            // 0..511 = 16 heads x 32 pairs
    int h = idx >> 5, j = idx & 31;
    size_t base = (size_t)row * (NHEAD * QD_DIM) + h * QD_DIM + 64;
    float a = q_buf[base + 2 * j];
    float b = q_buf[base + 2 * j + 1];
    o0[it] = a * cr[j] - b * sr[j];
    o1[it] = b * cr[j + 32] + a * sr[j + 32];
    hh[it] = h; jj[it] = j;
  }
  __syncthreads();                      // all reads before in-place writes
#pragma unroll
  for (int it = 0; it < 2; ++it) {
    size_t base = (size_t)row * (NHEAD * QD_DIM) + hh[it] * QD_DIM + 64;
    q_buf[base + jj[it]]      = o0[it];
    q_buf[base + 32 + jj[it]] = o1[it];
  }
  if (t < 32) {
    int j = t;
    float a = ckv_full[(size_t)row * 576 + 512 + 2 * j];
    float b = ckv_full[(size_t)row * 576 + 512 + 2 * j + 1];
    k_pe_rot[(size_t)row * 64 + j]      = a * cr[j] - b * sr[j];
    k_pe_rot[(size_t)row * 64 + 32 + j] = b * cr[j + 32] + a * sr[j + 32];
  }
}

// ---------------------------------------------------------------------------
// Flash attention (causal), bf16 WMMA. Block = 128 threads = 4 waves,
// each wave owns 16 query rows (BM=64), key block BN=64, head dim 128.
// Staging loads batched for MLP; V transpose via 4x4 register blocks.
// ---------------------------------------------------------------------------
#define AT_BM 64
#define AT_BN 64
#define AT_LQ 136   // 128 + 8 pad (ushorts)
#define AT_LV 72    // 64 + 8 pad

__global__ __launch_bounds__(128)
void mla_attention_kernel(const float* __restrict__ q_buf,
                          const float* __restrict__ kv_buf,
                          const float* __restrict__ k_pe,
                          float* __restrict__ attn_buf) {
  const int h    = blockIdx.y;
  const int m0   = blockIdx.x * AT_BM;
  const int t    = threadIdx.x;
  const int w    = t >> 5;
  const int lane = t & 31;
  const int l16  = lane & 15;
  const int half = lane >> 4;
  const int qrow = w * 16;

  __shared__ __align__(16) unsigned short Qs[AT_BM * AT_LQ];        // 17408 B
  __shared__ __align__(16) unsigned short Ks[AT_BN * AT_LQ];        // 17408 B
  __shared__ __align__(16) unsigned short Vts[128 * AT_LV];         // 18432 B
  __shared__ __align__(16) unsigned short Ps[4][16 * AT_LV];        //  9216 B

  // ---- stage Q (64 x 128), once; 8 loads in flight per batch ----
#pragma unroll
  for (int g = 0; g < 2; ++g) {
    float4 tmp[8];
#pragma unroll
    for (int i = 0; i < 8; ++i) {
      int linear = (t + (g * 8 + i) * 128) * 4;
      int r = linear >> 7, d = linear & 127;
      tmp[i] = *reinterpret_cast<const float4*>(
          &q_buf[(size_t)(m0 + r) * (NHEAD * QD_DIM) + h * QD_DIM + d]);
    }
#pragma unroll
    for (int i = 0; i < 8; ++i) {
      int linear = (t + (g * 8 + i) * 128) * 4;
      int r = linear >> 7, d = linear & 127;
      store4_bf16(&Qs[r * AT_LQ + d], tmp[i]);
    }
  }

  f32x8 o[8];
#pragma unroll
  for (int dt = 0; dt < 8; ++dt)
#pragma unroll
    for (int r = 0; r < 8; ++r) o[dt][r] = 0.0f;
  float m_run[8], l_run[8];
#pragma unroll
  for (int r = 0; r < 8; ++r) { m_run[r] = NEG_BIG; l_run[r] = 0.0f; }

  __syncthreads();

  const int kb_end = (m0 >> 6) + 1;   // causal: keys <= m0+63
  for (int kb = 0; kb < kb_end; ++kb) {
    const int n0 = kb * AT_BN;
    // ---- stage K_full (64 keys x 128): nope from kv_buf, pe shared ----
#pragma unroll
    for (int g = 0; g < 2; ++g) {
      float4 tmp[8];
#pragma unroll
      for (int i = 0; i < 8; ++i) {
        int linear = (t + (g * 8 + i) * 128) * 4;
        int j = linear >> 7, d = linear & 127;
        if (d < 64)
          tmp[i] = *reinterpret_cast<const float4*>(
              &kv_buf[(size_t)(n0 + j) * (NHEAD * KVD_DIM) + h * KVD_DIM + d]);
        else
          tmp[i] = *reinterpret_cast<const float4*>(
              &k_pe[(size_t)(n0 + j) * 64 + (d - 64)]);
      }
#pragma unroll
      for (int i = 0; i < 8; ++i) {
        int linear = (t + (g * 8 + i) * 128) * 4;
        int j = linear >> 7, d = linear & 127;
        store4_bf16(&Ks[j * AT_LQ + d], tmp[i]);
      }
    }
    // ---- stage V transposed via 4x4 register blocks: Vts[d][key] ----
#pragma unroll
    for (int i = 0; i < 4; ++i) {
      int bid = t + i * 128;            // 0..511 blocks of 4x4
      int dq = (bid & 31) * 4;          // 0..124
      int jq = (bid >> 5) * 4;          // 0..60
      float4 vv[4];
#pragma unroll
      for (int j = 0; j < 4; ++j)
        vv[j] = *reinterpret_cast<const float4*>(
            &kv_buf[(size_t)(n0 + jq + j) * (NHEAD * KVD_DIM) + h * KVD_DIM + 64 + dq]);
      store4_bf16v(&Vts[(dq + 0) * AT_LV + jq], vv[0].x, vv[1].x, vv[2].x, vv[3].x);
      store4_bf16v(&Vts[(dq + 1) * AT_LV + jq], vv[0].y, vv[1].y, vv[2].y, vv[3].y);
      store4_bf16v(&Vts[(dq + 2) * AT_LV + jq], vv[0].z, vv[1].z, vv[2].z, vv[3].z);
      store4_bf16v(&Vts[(dq + 3) * AT_LV + jq], vv[0].w, vv[1].w, vv[2].w, vv[3].w);
    }
    __syncthreads();

    // ---- S = Q . K^T  (16 x 64 per wave) ----
    bf16x16 afr[4];
#pragma unroll
    for (int kk = 0; kk < 4; ++kk) {
      const unsigned short* qb = &Qs[(qrow + l16) * AT_LQ + kk * 32];
      afr[kk] = load_frag(qb + half * 8, qb + 16 + half * 8);
    }
    f32x8 s[4];
#pragma unroll
    for (int nt = 0; nt < 4; ++nt) {
#pragma unroll
      for (int r = 0; r < 8; ++r) s[nt][r] = 0.0f;
#pragma unroll
      for (int kk = 0; kk < 4; ++kk) {
        const unsigned short* kbp = &Ks[(nt * 16 + l16) * AT_LQ + kk * 32];
        bf16x16 bfr = load_frag(kbp + half * 8, kbp + 16 + half * 8);
        s[nt] = wmma_bf16(afr[kk], bfr, s[nt]);
      }
    }

    // ---- scale + causal mask; per-row max (rows = 8*half + r) ----
    float mrow[8];
#pragma unroll
    for (int r = 0; r < 8; ++r) mrow[r] = NEG_BIG;
#pragma unroll
    for (int nt = 0; nt < 4; ++nt) {
      int col = n0 + nt * 16 + l16;
#pragma unroll
      for (int r = 0; r < 8; ++r) {
        int row = m0 + qrow + 8 * half + r;
        float v = s[nt][r] * ATT_SCALE;
        if (col > row) v = NEG_BIG;
        s[nt][r] = v;
        mrow[r] = fmaxf(mrow[r], v);
      }
    }
#pragma unroll
    for (int off = 1; off < 16; off <<= 1)
#pragma unroll
      for (int r = 0; r < 8; ++r)
        mrow[r] = fmaxf(mrow[r], __shfl_xor(mrow[r], off, 32));

    // ---- online softmax update ----
    float alpha[8];
#pragma unroll
    for (int r = 0; r < 8; ++r) {
      float newm = fmaxf(m_run[r], mrow[r]);
      alpha[r] = __expf(m_run[r] - newm);
      m_run[r] = newm;
    }
#pragma unroll
    for (int dt = 0; dt < 8; ++dt)
#pragma unroll
      for (int r = 0; r < 8; ++r) o[dt][r] *= alpha[r];

    float rsum[8];
#pragma unroll
    for (int r = 0; r < 8; ++r) rsum[r] = 0.0f;
#pragma unroll
    for (int nt = 0; nt < 4; ++nt)
#pragma unroll
      for (int r = 0; r < 8; ++r) {
        float p = __expf(s[nt][r] - m_run[r]);
        s[nt][r] = p;
        rsum[r] += p;
      }
#pragma unroll
    for (int off = 1; off < 16; off <<= 1)
#pragma unroll
      for (int r = 0; r < 8; ++r)
        rsum[r] += __shfl_xor(rsum[r], off, 32);
#pragma unroll
    for (int r = 0; r < 8; ++r) l_run[r] = l_run[r] * alpha[r] + rsum[r];

    // ---- P (C-layout) -> per-wave LDS tile (A-layout source) ----
#pragma unroll
    for (int nt = 0; nt < 4; ++nt)
#pragma unroll
      for (int r = 0; r < 8; ++r)
        Ps[w][(8 * half + r) * AT_LV + nt * 16 + l16] = f32_to_bf16(s[nt][r]);
    // in-wave RAW on LDS: CDNA5 split DS counter wait (no block barrier needed)
    asm volatile("s_wait_dscnt 0" ::: "memory");

    // ---- O += P . V ----
#pragma unroll
    for (int kk2 = 0; kk2 < 2; ++kk2) {
      const unsigned short* pb = &Ps[w][l16 * AT_LV + kk2 * 32];
      bf16x16 ap = load_frag(pb + half * 8, pb + 16 + half * 8);
#pragma unroll
      for (int dt = 0; dt < 8; ++dt) {
        const unsigned short* vb = &Vts[(dt * 16 + l16) * AT_LV + kk2 * 32];
        bf16x16 bv = load_frag(vb + half * 8, vb + 16 + half * 8);
        o[dt] = wmma_bf16(ap, bv, o[dt]);
      }
    }
    __syncthreads();   // protect Ks/Vts before next stage
  }

  // ---- normalize + write ----
  float inv[8];
#pragma unroll
  for (int r = 0; r < 8; ++r) inv[r] = 1.0f / l_run[r];
#pragma unroll
  for (int dt = 0; dt < 8; ++dt)
#pragma unroll
    for (int r = 0; r < 8; ++r) {
      int row = m0 + qrow + 8 * half + r;
      attn_buf[(size_t)row * (NHEAD * 128) + h * 128 + dt * 16 + l16] =
          o[dt][r] * inv[r];
    }
}

// ---------------------------------------------------------------------------
// Launch: q-proj, kv_a-proj, rmsnorm, rope, kv_b-proj, attention, o-proj
// ---------------------------------------------------------------------------
extern "C" void kernel_launch(void* const* d_in, const int* in_sizes, int n_in,
                              void* d_out, int out_size, void* d_ws, size_t ws_size,
                              hipStream_t stream) {
  const float* hidden  = (const float*)d_in[0];
  const float* cosb    = (const float*)d_in[1];
  const float* sinb    = (const float*)d_in[2];
  const float* w_q     = (const float*)d_in[3];
  const float* w_kv_a  = (const float*)d_in[4];
  const float* kv_ln   = (const float*)d_in[5];
  const float* w_kv_b  = (const float*)d_in[6];
  const float* w_o     = (const float*)d_in[7];
  float* out = (float*)d_out;

  float* ws       = (float*)d_ws;
  float* q_buf    = ws;                                  // 2048*2048
  float* ckv_full = q_buf    + (size_t)S_LEN * 2048;     // 2048*576
  float* ckv_norm = ckv_full + (size_t)S_LEN * 576;      // 2048*512
  float* k_pe_rot = ckv_norm + (size_t)S_LEN * 512;      // 2048*64
  float* kv_buf   = k_pe_rot + (size_t)S_LEN * 64;       // 2048*3072
  float* attn_buf = kv_buf   + (size_t)S_LEN * 3072;     // 2048*2048

  dim3 blk256(256), blk128(128);

  // q = hidden @ w_q            [2048,2048]x[2048,2048]
  gemm_bf16_kernel<<<dim3(16, 16), blk256, 0, stream>>>(
      hidden, w_q, q_buf, S_LEN, 2048, 2048);
  // ckv_full = hidden @ w_kv_a  [2048,2048]x[2048,576]
  gemm_bf16_kernel<<<dim3(5, 16), blk256, 0, stream>>>(
      hidden, w_kv_a, ckv_full, S_LEN, 576, 2048);
  // rmsnorm latent
  rmsnorm_kernel<<<S_LEN, blk256, 0, stream>>>(ckv_full, kv_ln, ckv_norm);
  // rope q_pe (in-place) + k_pe
  rope_kernel<<<S_LEN, blk256, 0, stream>>>(q_buf, ckv_full, k_pe_rot, cosb, sinb);
  // kv = ckv_norm @ w_kv_b      [2048,512]x[512,3072]
  gemm_bf16_kernel<<<dim3(24, 16), blk256, 0, stream>>>(
      ckv_norm, w_kv_b, kv_buf, S_LEN, 3072, 512);
  // flash attention per head
  mla_attention_kernel<<<dim3(S_LEN / AT_BM, NHEAD), blk128, 0, stream>>>(
      q_buf, kv_buf, k_pe_rot, attn_buf);
  // out = attn @ w_o            [2048,2048]x[2048,2048]
  gemm_bf16_kernel<<<dim3(16, 16), blk256, 0, stream>>>(
      attn_buf, w_o, out, S_LEN, 2048, 2048);
}
